// SparseConvBackbone_39127152067017
// MI455X (gfx1250) — matrix operand order repaired and checked by reading, not compiled, not run on hardware
//
#include <hip/hip_runtime.h>

#define BSZ   4
#define NB    16384
#define SEEDN 128
#define COUTF 192
#define NPTS  20000

typedef __attribute__((ext_vector_type(16))) __bf16         v16bf;
typedef __attribute__((ext_vector_type(8)))  float          v8f;
typedef __attribute__((ext_vector_type(16))) unsigned short v16us;
typedef __attribute__((ext_vector_type(8)))  unsigned short v8us;

static __device__ __forceinline__ unsigned short f32_to_bf16(float f) {
  unsigned int u = __builtin_bit_cast(unsigned int, f);
  u += 0x7FFFu + ((u >> 16) & 1u);   // round-to-nearest-even
  return (unsigned short)(u >> 16);
}
static __device__ __forceinline__ float bf16_to_f32(unsigned short h) {
  unsigned int u = (unsigned int)h << 16;
  return __builtin_bit_cast(float, u);
}
static __device__ __forceinline__ v16us ldfrag(const unsigned short* p, int hk) {
  v8us lo = *(const v8us*)(p + hk);
  v8us hi = *(const v8us*)(p + 16 + hk);
  return __builtin_shufflevector(lo, hi, 0,1,2,3,4,5,6,7,8,9,10,11,12,13,14,15);
}

// ---------------- elementwise / utility kernels ----------------

__global__ void k_zero(float* __restrict__ p, long n) {
  long i = (long)blockIdx.x * blockDim.x + threadIdx.x;
  if (i < n) p[i] = 0.0f;
}

__global__ void k_cvt_bf16(const float* __restrict__ s, unsigned short* __restrict__ d, long n) {
  long i = (long)blockIdx.x * blockDim.x + threadIdx.x;
  if (i < n) d[i] = f32_to_bf16(s[i]);
}

// Pack f32 weights [K][Cin][Cout] into WMMA B-fragment order:
// frag f = (ko*ntiles + nt)*nchunks + c32 ; within frag: [lane][16 halves].
__global__ void k_pack_w(const float* __restrict__ src, unsigned short* __restrict__ dst,
                         int Cin, int Cout, int nchunks, int ntiles, long tot) {
  long o = (long)blockIdx.x * blockDim.x + threadIdx.x;
  if (o >= tot) return;
  int ii   = (int)(o & 15);
  int lane = (int)((o >> 4) & 31);
  long f   = o >> 9;
  int c32  = (int)(f % nchunks);
  long r   = f / nchunks;
  int nt   = (int)(r % ntiles);
  int ko   = (int)(r / ntiles);
  int n    = lane & 15, grp = lane >> 4;
  int kk   = (ii & 7) + (grp << 3) + ((ii >> 3) << 4);
  int k    = (c32 << 5) + kk;
  int col  = (nt << 4) + n;
  unsigned short v = 0;
  if (k < Cin) v = f32_to_bf16(src[((long)ko * Cin + k) * Cout + col]);
  dst[o] = v;
}

__global__ void k_bn_stats(const float* __restrict__ x, int n, int C,
                           float* __restrict__ mean, float* __restrict__ rstd) {
  int c = blockIdx.x;
  float s = 0.f, s2 = 0.f;
  for (int i = threadIdx.x; i < n; i += blockDim.x) {
    float v = x[(long)i * C + c];
    s += v; s2 += v * v;
  }
  __shared__ float sh[256], sh2[256];
  sh[threadIdx.x] = s; sh2[threadIdx.x] = s2;
  __syncthreads();
  for (int o = 128; o > 0; o >>= 1) {
    if (threadIdx.x < o) { sh[threadIdx.x] += sh[threadIdx.x + o]; sh2[threadIdx.x] += sh2[threadIdx.x + o]; }
    __syncthreads();
  }
  if (threadIdx.x == 0) {
    float mu = sh[0] / (float)n;
    float var = sh2[0] / (float)n - mu * mu;
    mean[c] = mu;
    rstd[c] = rsqrtf(var + 1e-5f);
  }
}

// BN apply: f32 conv accumulator in, bf16 activation out
__global__ void k_bn_apply(const float* __restrict__ x, const float* __restrict__ mean,
                           const float* __restrict__ rstd, const float* __restrict__ g,
                           const float* __restrict__ b, unsigned short* __restrict__ y,
                           long tot, int C, int relu) {
  long i = (long)blockIdx.x * blockDim.x + threadIdx.x;
  if (i >= tot) return;
  int c = (int)(i % C);
  float v = (x[i] - mean[c]) * rstd[c] * g[c] + b[c];
  if (relu) v = fmaxf(v, 0.0f);
  y[i] = f32_to_bf16(v);
}

__global__ void k_add_relu(const unsigned short* __restrict__ a, const unsigned short* __restrict__ b,
                           unsigned short* __restrict__ y, long n) {
  long i = (long)blockIdx.x * blockDim.x + threadIdx.x;
  if (i < n) y[i] = f32_to_bf16(fmaxf(bf16_to_f32(a[i]) + bf16_to_f32(b[i]), 0.0f));
}

__global__ void k_concat_relu_a(const unsigned short* __restrict__ a, int Ca,
                                const unsigned short* __restrict__ b, int Cb,
                                unsigned short* __restrict__ y, long n) {
  int C = Ca + Cb;
  long tot = n * (long)C;
  long i = (long)blockIdx.x * blockDim.x + threadIdx.x;
  if (i >= tot) return;
  long r = i / C; int c = (int)(i % C);
  unsigned short v;
  if (c < Ca) { v = a[r * Ca + c]; if (v & 0x8000u) v = 0; }  // bf16 relu
  else        v = b[r * Cb + (c - Ca)];
  y[i] = v;
}

__global__ void k_rownorm(const float* __restrict__ F, long n, int C, float* __restrict__ out) {
  long i = (long)blockIdx.x * blockDim.x + threadIdx.x;
  if (i >= n) return;
  float s = 0.f;
  const float* row = F + i * C;
  for (int c = 0; c < C; ++c) s += row[c] * row[c];
  out[i] = sqrtf(s);
}

__global__ void k_topk(const float* __restrict__ norms, float* __restrict__ work,
                       int nb, int S, int* __restrict__ outIdx) {
  int b = blockIdx.x;
  const float* src = norms + (long)b * nb;
  float* w = work + (long)b * nb;
  for (int i = threadIdx.x; i < nb; i += blockDim.x) w[i] = src[i];
  __syncthreads();
  __shared__ float sv[256];
  __shared__ int   si[256];
  for (int s = 0; s < S; ++s) {
    float best = -3.4e38f; int bi = 0;
    for (int i = threadIdx.x; i < nb; i += blockDim.x) {
      float v = w[i];
      if (v > best) { best = v; bi = i; }
    }
    sv[threadIdx.x] = best; si[threadIdx.x] = bi;
    __syncthreads();
    for (int o = 128; o > 0; o >>= 1) {
      if (threadIdx.x < o && sv[threadIdx.x + o] > sv[threadIdx.x]) {
        sv[threadIdx.x] = sv[threadIdx.x + o]; si[threadIdx.x] = si[threadIdx.x + o];
      }
      __syncthreads();
    }
    if (threadIdx.x == 0) { outIdx[b * S + s] = si[0]; w[si[0]] = -3.4e38f; }
    __syncthreads();
  }
}

__global__ void k_gather_out(const float* __restrict__ F, const int* __restrict__ sel,
                             const float* __restrict__ points, const int* __restrict__ inds,
                             const int* __restrict__ coords, float* __restrict__ dout) {
  int t = blockIdx.x * blockDim.x + threadIdx.x;
  if (t >= BSZ * SEEDN) return;
  int b = t / SEEDN, s = t % SEEDN;
  int vi = sel[t];
  long row = (long)b * NB + vi;
  int pi = inds[row];
  float* oxyz  = dout;
  float* ofeat = dout + (long)BSZ * SEEDN * 3;
  int*   oind  = (int*)(dout + (long)BSZ * SEEDN * 3 + (long)BSZ * COUTF * SEEDN);
  int*   ocrd  = oind + BSZ * SEEDN;
  for (int j = 0; j < 3; ++j) oxyz[(long)t * 3 + j] = points[((long)b * NPTS + pi) * 3 + j];
  for (int c = 0; c < COUTF; ++c)
    ofeat[((long)b * COUTF + c) * SEEDN + s] = F[row * COUTF + c];
  oind[t] = pi;
  for (int j = 0; j < 4; ++j) ocrd[(long)t * 4 + j] = coords[row * 4 + j];
}

// ---------------- WMMA kernels ----------------
// Sparse conv on bf16 activations. Staging: zero LDS tile, barrier, then
// async DMA gather (global_load_async_to_lds_b128, per-lane addresses) of the
// valid 16B row-chunks, s_wait_asynccnt + barrier. GEMM: each wave owns an
// N-tile and runs BOTH M-tiles against the shared B fragment with two
// independent accumulators (hides the 5-NOP WMMA->WMMA RAW hazard).
__global__ __launch_bounds__(256)
void k_sconv_wmma(const unsigned short* __restrict__ x, const unsigned short* __restrict__ wpk,
                  const int* __restrict__ in_idx, const int* __restrict__ out_idx,
                  const unsigned char* __restrict__ mask,
                  int M, int Cin, int cinLog2, int Cout, float* __restrict__ out) {
  const int CinPad  = 1 << cinLog2;
  const int nchunks = CinPad >> 5;
  const int ntiles  = Cout >> 4;
  const int ko      = blockIdx.y;
  const long koM    = (long)ko * M;
  const int mbase   = blockIdx.x * 32;

  extern __shared__ v16us AsV[];
  unsigned short* As = (unsigned short*)AsV;

  // zero tile (covers masked rows + pad columns), then DMA the valid data in
  for (int i = threadIdx.x; i < (CinPad << 5); i += 256) As[i] = 0;
  __syncthreads();

  if ((Cin & 7) == 0) {
    const int cpr = Cin >> 3;              // 16-byte chunks per row
    const int total = 32 * cpr;
    for (int i = threadIdx.x; i < total; i += 256) {
      int r = i / cpr, cc = i - r * cpr;
      int m = mbase + r;
      if (m < M && mask[koM + m]) {
        int srow = in_idx[koM + m];
        unsigned lds = (unsigned)(size_t)(As + ((r << cinLog2) + (cc << 3)));
        unsigned long long ga = (unsigned long long)(size_t)(x + (long)srow * Cin + (cc << 3));
        asm volatile("global_load_async_to_lds_b128 %0, %1, off" :: "v"(lds), "v"(ga) : "memory");
      }
    }
    asm volatile("s_wait_asynccnt 0x0" ::: "memory");
  } else {                                  // Cin==3 (conv1) fallback
    for (int i = threadIdx.x; i < (CinPad << 5); i += 256) {
      int r = i >> cinLog2;
      int c = i & (CinPad - 1);
      int m = mbase + r;
      if (m < M && c < Cin && mask[koM + m]) {
        int srow = in_idx[koM + m];
        As[i] = x[(long)srow * Cin + c];
      }
    }
  }
  __syncthreads();

  const int lane = threadIdx.x & 31;
  const int wave = threadIdx.x >> 5;
  const int r16  = lane & 15;
  const int hk   = (lane >> 4) << 3;
  const v16us* wV = (const v16us*)wpk;

  for (int nt = wave; nt < ntiles; nt += 8) {
    const unsigned short* a0 = As + (r16 << cinLog2);
    const unsigned short* a1 = As + ((16 + r16) << cinLog2);
    const v16us* bp = wV + ((long)(ko * ntiles + nt) * nchunks << 5) + lane;
    v8f acc0 = {}, acc1 = {};
    for (int c32 = 0; c32 < nchunks; ++c32) {
      const int kc = c32 << 5;
      v16us au0 = ldfrag(a0 + kc, hk);
      v16us au1 = ldfrag(a1 + kc, hk);
      v16us bu  = bp[(long)c32 << 5];
      if (c32 + 1 < nchunks) __builtin_prefetch((const void*)(bp + ((long)(c32 + 1) << 5)), 0, 1);
      acc0 = __builtin_amdgcn_wmma_f32_16x16x32_bf16(
          false, __builtin_bit_cast(v16bf, au0), false, __builtin_bit_cast(v16bf, bu),
          (short)0, acc0, false, false);
      acc1 = __builtin_amdgcn_wmma_f32_16x16x32_bf16(
          false, __builtin_bit_cast(v16bf, au1), false, __builtin_bit_cast(v16bf, bu),
          (short)0, acc1, false, false);
    }
    const int col = (nt << 4) + r16;
#pragma unroll
    for (int j = 0; j < 8; ++j) {
      int m = mbase + j + ((lane >> 4) << 3);
      if (m < M && mask[koM + m]) {
        unsafeAtomicAdd(&out[(long)out_idx[koM + m] * Cout + col], acc0[j]);
      }
    }
#pragma unroll
    for (int j = 0; j < 8; ++j) {
      int m = mbase + 16 + j + ((lane >> 4) << 3);
      if (m < M && mask[koM + m]) {
        unsafeAtomicAdd(&out[(long)out_idx[koM + m] * Cout + col], acc1[j]);
      }
    }
  }
}

// Dense GEMM on bf16 input (1x1 conv / final linear). Async row staging,
// dual-M-tile accumulators, optional bf16 or f32 output.
__global__ __launch_bounds__(256)
void k_dense_wmma(const unsigned short* __restrict__ x, const unsigned short* __restrict__ wpk,
                  const float* __restrict__ bias, int n, int Cin, int Cout,
                  float* __restrict__ outF, unsigned short* __restrict__ outH, int relu) {
  const int nchunks = Cin >> 5;              // Cin in {96, 64}
  const int ntiles  = Cout >> 4;
  const int mbase   = blockIdx.x * 32;
  extern __shared__ v16us AsV[];
  unsigned short* As = (unsigned short*)AsV;

  for (int i = threadIdx.x; i < 32 * Cin; i += 256) As[i] = 0;
  __syncthreads();
  {
    const int cpr = Cin >> 3;
    const int total = 32 * cpr;
    for (int i = threadIdx.x; i < total; i += 256) {
      int r = i / cpr, cc = i - r * cpr;
      int m = mbase + r;
      if (m < n) {
        unsigned lds = (unsigned)(size_t)(As + (r * Cin + (cc << 3)));
        unsigned long long ga = (unsigned long long)(size_t)(x + (long)m * Cin + (cc << 3));
        asm volatile("global_load_async_to_lds_b128 %0, %1, off" :: "v"(lds), "v"(ga) : "memory");
      }
    }
    asm volatile("s_wait_asynccnt 0x0" ::: "memory");
  }
  __syncthreads();

  const int lane = threadIdx.x & 31;
  const int wave = threadIdx.x >> 5;
  const int r16  = lane & 15;
  const int hk   = (lane >> 4) << 3;
  const v16us* wV = (const v16us*)wpk;

  for (int nt = wave; nt < ntiles; nt += 8) {
    const unsigned short* a0 = As + r16 * Cin;
    const unsigned short* a1 = As + (16 + r16) * Cin;
    const v16us* bp = wV + ((long)nt * ((1 << (32 - __builtin_clz(Cin - 1))) >> 5) << 5) + lane;
    v8f acc0 = {}, acc1 = {};
    for (int c32 = 0; c32 < nchunks; ++c32) {
      const int kc = c32 << 5;
      v16us au0 = ldfrag(a0 + kc, hk);
      v16us au1 = ldfrag(a1 + kc, hk);
      v16us bu  = bp[(long)c32 << 5];
      acc0 = __builtin_amdgcn_wmma_f32_16x16x32_bf16(
          false, __builtin_bit_cast(v16bf, au0), false, __builtin_bit_cast(v16bf, bu),
          (short)0, acc0, false, false);
      acc1 = __builtin_amdgcn_wmma_f32_16x16x32_bf16(
          false, __builtin_bit_cast(v16bf, au1), false, __builtin_bit_cast(v16bf, bu),
          (short)0, acc1, false, false);
    }
    const int col = (nt << 4) + r16;
    float bb = bias ? bias[col] : 0.0f;
#pragma unroll
    for (int j = 0; j < 8; ++j) {
      int m0 = mbase + j + ((lane >> 4) << 3);
      int m1 = m0 + 16;
      float v0 = acc0[j] + bb, v1 = acc1[j] + bb;
      if (relu) { v0 = fmaxf(v0, 0.0f); v1 = fmaxf(v1, 0.0f); }
      if (m0 < n) { if (outH) outH[(long)m0 * Cout + col] = f32_to_bf16(v0); else outF[(long)m0 * Cout + col] = v0; }
      if (m1 < n) { if (outH) outH[(long)m1 * Cout + col] = f32_to_bf16(v1); else outF[(long)m1 * Cout + col] = v1; }
    }
  }
}

// ---------------- host orchestration ----------------

extern "C" void kernel_launch(void* const* d_in, const int* in_sizes, int n_in,
                              void* d_out, int out_size, void* d_ws, size_t ws_size,
                              hipStream_t stream) {
  (void)n_in; (void)out_size; (void)ws_size;
  auto F32 = [&](int i) { return (const float*)d_in[i]; };
  auto I32 = [&](int i) { return (const int*)d_in[i]; };
  auto U8  = [&](int i) { return (const unsigned char*)d_in[i]; };

  const long n0 = in_sizes[2] / 3;     // feats (n0, 3)
  const long n1 = in_sizes[71] / 4;
  const long n2 = in_sizes[72] / 4;
  const long n3 = in_sizes[73] / 4;

  char* wsb = (char*)d_ws;
  size_t wso = 0;
  auto alloc  = [&](size_t bytes) -> void* {
    size_t cur = (wso + 255) & ~(size_t)255;
    wso = cur + bytes;
    return (void*)(wsb + cur);
  };
  auto allocF = [&](long n) { return (float*)alloc((size_t)n * 4); };
  auto allocH = [&](long n) { return (unsigned short*)alloc((size_t)n * 2); };
  auto allocI = [&](long n) { return (int*)alloc((size_t)n * 4); };

  auto clog2 = [](int cin) { int l = 5; while ((1 << l) < cin) ++l; return l; };

  auto packw = [&](int idx, int K, int Cin, int Cout) -> const unsigned short* {
    int cinPad = 1 << clog2(Cin);
    int nch = cinPad >> 5, ntl = Cout >> 4;
    long tot = (long)K * nch * ntl * 512;
    unsigned short* p = allocH(tot);
    k_pack_w<<<dim3((unsigned)((tot + 255) / 256)), dim3(256), 0, stream>>>(
        F32(idx), p, Cin, Cout, nch, ntl, tot);
    return p;
  };
  auto zero = [&](float* p, long n) {
    k_zero<<<dim3((unsigned)((n + 255) / 256)), dim3(256), 0, stream>>>(p, n);
  };
  auto sconv = [&](const unsigned short* x, const unsigned short* w, int kidx, int K,
                   int Cin, int Cout, long nout, float* out) {
    long M = in_sizes[kidx] / K;
    zero(out, nout * Cout);
    int lg = clog2(Cin);
    dim3 grid((unsigned)((M + 31) / 32), (unsigned)K);
    size_t sh = (size_t)32 * (1 << lg) * 2;
    k_sconv_wmma<<<grid, dim3(256), sh, stream>>>(x, w, I32(kidx), I32(kidx + 1), U8(kidx + 2),
                                                  (int)M, Cin, lg, Cout, out);
  };

  float* mean = allocF(256);
  float* rstd = allocF(256);
  auto bn = [&](const float* x, int gidx, long n, int C, unsigned short* y, int relu) {
    k_bn_stats<<<dim3((unsigned)C), dim3(256), 0, stream>>>(x, (int)n, C, mean, rstd);
    long tot = n * C;
    k_bn_apply<<<dim3((unsigned)((tot + 255) / 256)), dim3(256), 0, stream>>>(
        x, mean, rstd, F32(gidx), F32(gidx + 1), y, tot, C, relu);
  };
  auto addrelu = [&](const unsigned short* a, const unsigned short* b, unsigned short* y, long n) {
    k_add_relu<<<dim3((unsigned)((n + 255) / 256)), dim3(256), 0, stream>>>(a, b, y, n);
  };
  auto concat = [&](const unsigned short* a, int Ca, const unsigned short* b, int Cb,
                    unsigned short* y, long n) {
    long tot = n * (Ca + Cb);
    k_concat_relu_a<<<dim3((unsigned)((tot + 255) / 256)), dim3(256), 0, stream>>>(a, Ca, b, Cb, y, n);
  };
  auto dense = [&](const unsigned short* x, const unsigned short* w, const float* bias,
                   long n, int Cin, int Cout, float* yF, unsigned short* yH, int relu) {
    size_t sh = (size_t)32 * Cin * 2;
    k_dense_wmma<<<dim3((unsigned)((n + 31) / 32)), dim3(256), sh, stream>>>(
        x, w, bias, (int)n, Cin, Cout, yF, yH, relu);
  };
  // residual block: pbase = index of w1 (n1.g at pbase+1, n2.g at pbase+4)
  auto blockfn = [&](const unsigned short* x, int pbase, int kidx, long n, int C,
                     float* accum, unsigned short* hA, unsigned short* hB, unsigned short* dst,
                     const unsigned short* w1, const unsigned short* w2) {
    sconv(x, w1, kidx, 27, C, C, n, accum);
    bn(accum, pbase + 1, n, C, hA, 1);
    sconv(hA, w2, kidx, 27, C, C, n, accum);
    bn(accum, pbase + 4, n, C, hB, 0);
    addrelu(hB, x, dst, n * C);
  };

  // --- weight packing ---
  const unsigned short* wConv1  = packw(5, 125, 3, 32);
  const unsigned short* wB1a    = packw(8, 27, 32, 32);
  const unsigned short* wB1b    = packw(11, 27, 32, 32);
  const unsigned short* wConv2  = packw(14, 27, 32, 64);
  const unsigned short* wB2a    = packw(17, 27, 64, 64);
  const unsigned short* wB2b    = packw(20, 27, 64, 64);
  const unsigned short* wConv3  = packw(23, 27, 64, 128);
  const unsigned short* wB3a    = packw(26, 27, 128, 128);
  const unsigned short* wB3b    = packw(29, 27, 128, 128);
  const unsigned short* wConv4  = packw(32, 27, 128, 256);
  const unsigned short* wB4a    = packw(35, 27, 256, 256);
  const unsigned short* wB4b    = packw(38, 27, 256, 256);
  const unsigned short* wConv4t = packw(41, 27, 256, 128);
  const unsigned short* wB4ta   = packw(44, 27, 128, 128);
  const unsigned short* wB4tb   = packw(47, 27, 128, 128);
  const unsigned short* wConv3t = packw(50, 27, 256, 64);
  const unsigned short* wB3ta   = packw(53, 27, 64, 64);
  const unsigned short* wB3tb   = packw(56, 27, 64, 64);
  const unsigned short* wConv2t = packw(59, 27, 128, 64);
  const unsigned short* wB2ta   = packw(62, 27, 64, 64);
  const unsigned short* wB2tb   = packw(65, 27, 64, 64);
  const unsigned short* wConv1t = packw(68, 1, 96, 64);
  const unsigned short* wFinal  = packw(69, 1, 64, 192);

  // --- buffers ---
  unsigned short* fb = allocH(n0 * 3);                       // bf16 feats
  float* a0f = allocF(n0 * 64);                              // f32 conv accumulators
  float* a1f = allocF(n1 * 64);
  float* a2f = allocF(n2 * 128);
  float* a3f = allocF(n3 * 256);
  unsigned short* xb0 = allocH(n0 * 32);
  unsigned short* h0a = allocH(n0 * 32); unsigned short* h0b = allocH(n0 * 32);
  unsigned short* s1  = allocH(n0 * 32);
  unsigned short* xb1 = allocH(n1 * 64);
  unsigned short* h1a = allocH(n1 * 64); unsigned short* h1b = allocH(n1 * 64);
  unsigned short* s2  = allocH(n1 * 64);
  unsigned short* xb2 = allocH(n2 * 128);
  unsigned short* h2a = allocH(n2 * 128); unsigned short* h2b = allocH(n2 * 128);
  unsigned short* s4  = allocH(n2 * 128);
  unsigned short* xb3 = allocH(n3 * 256);
  unsigned short* h3a = allocH(n3 * 256); unsigned short* h3b = allocH(n3 * 256);
  unsigned short* s8  = allocH(n3 * 256);
  unsigned short* d2  = allocH(n2 * 128); unsigned short* cat2 = allocH(n2 * 256);
  unsigned short* d1  = allocH(n1 * 64);  unsigned short* cat1 = allocH(n1 * 128);
  unsigned short* x0d = allocH(n0 * 64);
  unsigned short* h0c = allocH(n0 * 64);  unsigned short* h0d = allocH(n0 * 64);
  unsigned short* d0  = allocH(n0 * 64);  unsigned short* cat0 = allocH(n0 * 96);
  unsigned short* h64 = allocH(n0 * 64);
  float* Fb  = allocF(n0 * 192);
  float* nrm = allocF(n0);
  float* nwk = allocF(n0);
  int*   sel = allocI((long)BSZ * SEEDN);

  k_cvt_bf16<<<dim3((unsigned)((n0 * 3 + 255) / 256)), dim3(256), 0, stream>>>(F32(2), fb, n0 * 3);

  // --- encoder ---
  sconv(fb, wConv1, 74, 125, 3, 32, n0, a0f);               // conv1, k0_5
  bn(a0f, 6, n0, 32, xb0, 0);                               // bn1
  blockfn(xb0, 8, 77, n0, 32, a0f, h0a, h0b, s1, wB1a, wB1b);  // block1, k0_3

  sconv(s1, wConv2, 80, 27, 32, 64, n1, a1f);               // conv2, k01
  bn(a1f, 15, n1, 64, xb1, 0);                              // bn2
  blockfn(xb1, 17, 83, n1, 64, a1f, h1a, h1b, s2, wB2a, wB2b); // block2, k1_3

  sconv(s2, wConv3, 86, 27, 64, 128, n2, a2f);              // conv3, k12
  bn(a2f, 24, n2, 128, xb2, 0);                             // bn3
  blockfn(xb2, 26, 89, n2, 128, a2f, h2a, h2b, s4, wB3a, wB3b); // block3, k2_3

  sconv(s4, wConv4, 92, 27, 128, 256, n3, a3f);             // conv4, k23
  bn(a3f, 33, n3, 256, xb3, 0);                             // bn4
  blockfn(xb3, 35, 95, n3, 256, a3f, h3a, h3b, s8, wB4a, wB4b); // block4, k3_3

  // --- decoder ---
  sconv(s8, wConv4t, 98, 27, 256, 128, n2, a2f);            // conv4_tr, k32
  bn(a2f, 42, n2, 128, xb2, 0);                             // bn4_tr
  blockfn(xb2, 44, 89, n2, 128, a2f, h2a, h2b, d2, wB4ta, wB4tb); // block4_tr, k2_3
  concat(d2, 128, s4, 128, cat2, n2);

  sconv(cat2, wConv3t, 101, 27, 256, 64, n1, a1f);          // conv3_tr, k21
  bn(a1f, 51, n1, 64, xb1, 0);                              // bn3_tr
  blockfn(xb1, 53, 83, n1, 64, a1f, h1a, h1b, d1, wB3ta, wB3tb);  // block3_tr, k1_3
  concat(d1, 64, s2, 64, cat1, n1);

  sconv(cat1, wConv2t, 104, 27, 128, 64, n0, a0f);          // conv2_tr, k10
  bn(a0f, 60, n0, 64, x0d, 0);                              // bn2_tr
  blockfn(x0d, 62, 77, n0, 64, a0f, h0c, h0d, d0, wB2ta, wB2tb);  // block2_tr, k0_3
  concat(d0, 64, s1, 32, cat0, n0);

  dense(cat0, wConv1t, nullptr, n0, 96, 64, nullptr, h64, 1);   // conv1_tr + relu -> bf16
  dense(h64, wFinal, F32(70), n0, 64, 192, Fb, nullptr, 0);     // final -> f32

  // --- seed selection + output gathers ---
  k_rownorm<<<dim3((unsigned)((n0 + 255) / 256)), dim3(256), 0, stream>>>(Fb, n0, COUTF, nrm);
  k_topk<<<dim3(BSZ), dim3(256), 0, stream>>>(nrm, nwk, NB, SEEDN, sel);
  k_gather_out<<<dim3((BSZ * SEEDN + 63) / 64), dim3(64), 0, stream>>>(
      Fb, sel, F32(0), I32(3), I32(1), (float*)d_out);
}
